// LocalSparseAttention_42717744726488
// MI455X (gfx1250) — matrix operand order repaired and checked by reading, not compiled, not run on hardware
//
#include <hip/hip_runtime.h>

typedef _Float16 h4   __attribute__((ext_vector_type(4)));
typedef _Float16 h8   __attribute__((ext_vector_type(8)));
typedef _Float16 v16h __attribute__((ext_vector_type(16)));
typedef float    v8f  __attribute__((ext_vector_type(8)));
typedef unsigned int v4u __attribute__((ext_vector_type(4)));
typedef int      v8i_t __attribute__((ext_vector_type(8)));

#define WMMA_F16(a, b, c) \
  __builtin_amdgcn_wmma_f32_16x16x32_f16(false, (a), false, (b), (short)0, (c), false, false)

// A-fragment (16x32 f16): lane l holds row m=l%16; K chunks {kb..kb+7, kb+16..kb+23}, kb = (l<16)?0:8
__device__ __forceinline__ v16h frag_a(const _Float16* base, int stride, int row, int col, int lane) {
  const int m  = lane & 15;
  const int kb = (lane & 16) ? 8 : 0;
  const _Float16* p = base + (row + m) * stride + col + kb;
  h8 lo = *(const h8*)p;
  h8 hi = *(const h8*)(p + 16);
  return __builtin_shufflevector(lo, hi, 0,1,2,3,4,5,6,7,8,9,10,11,12,13,14,15);
}

// B-fragment (32x16 f16) loaded from B^T stored row-major [n][k]:
// lane l holds col n=l%16; K chunks {kb..kb+7, kb+16..kb+23}, kb = (l<16)?0:8
__device__ __forceinline__ v16h frag_b(const _Float16* baseT, int stride, int ncol, int kcol, int lane) {
  const int n  = lane & 15;
  const int kb = (lane & 16) ? 8 : 0;
  const _Float16* p = baseT + (ncol + n) * stride + kcol + kb;
  h8 lo = *(const h8*)p;
  h8 hi = *(const h8*)(p + 16);
  return __builtin_shufflevector(lo, hi, 0,1,2,3,4,5,6,7,8,9,10,11,12,13,14,15);
}

// TDM: DMA a 2D tile (32 rows x 512B) global -> LDS, inserting 16B of LDS
// padding every 512B (row stride 528B = 264 halves). data_size=8B units.
// D# packing per CDNA5 ISA ch.8.3/8.4.
__device__ __forceinline__ void tdm_load_wblock(unsigned lds_byte_addr,
                                                unsigned long long gaddr) {
  v4u g0;
  g0.x = 1u;                                  // count=1, user descriptor, no gather
  g0.y = lds_byte_addr;                       // lds_addr [63:32]
  g0.z = (unsigned)gaddr;                     // global_addr [95:64]
  g0.w = (unsigned)((gaddr >> 32) & 0x1FFFFFFull) | 0x80000000u; // addr[120:96], type=2
  v8i_t g1;
  g1[0] = (3 << 16)        // data_size = 3 (8 bytes)
        | (1 << 20)        // pad_enable
        | (6 << 22)        // pad_interval: 64 x 8B = 512B
        | (3 << 25);       // pad_amount: 4 DWORDs = 16B
  g1[1] = (64 << 16);      // tensor_dim0 lo16 = 64 units (512B row)
  g1[2] = (32 << 16);      // tensor_dim0 hi16 = 0 ; tensor_dim1 lo16 = 32 rows
  g1[3] = (64 << 16);      // tensor_dim1 hi16 = 0 ; tile_dim0 = 64
  g1[4] = 32;              // tile_dim1 = 32 ; tile_dim2 = 0 (2D)
  g1[5] = 64;              // tensor_dim0_stride lo32 = 64 units
  g1[6] = 0;               // stride0 hi16 = 0 ; tensor_dim1_stride lo16 = 0
  g1[7] = 0;
  asm volatile("tensor_load_to_lds %0, %1" :: "s"(g0), "s"(g1) : "memory");
}

// ---------------- kernel 0: transpose + f32->f16 weights -----------------
__global__ __launch_bounds__(256) void prep_weights_kernel(
    const float* __restrict__ qkv_w, const float* __restrict__ proj_w,
    _Float16* __restrict__ wqkvT, _Float16* __restrict__ wpT) {
  const int row = blockIdx.x;     // 0..1023
  const int c   = threadIdx.x;    // 0..255
  if (row < 768) {
    wqkvT[(row << 8) + c] = (_Float16)qkv_w[c * 768 + row];
  } else {
    const int n = row - 768;
    wpT[(n << 8) + c] = (_Float16)proj_w[(c << 8) + n];
  }
}

// ---------------- kernel 1: fused per-window QKV + attention -------------
// LDS layout (halves):
//   lx  @      0 : [256][264] window x (f16), persistent
//   lq  @  67584 : [256][40]  Q   (phase 1-2)     \
//   lk  @  77824 : [256][40]  K   (phase 1-2)      > all overlaid by
//   lw  @  88064 : [96][264]  W^T (phase 1)       /  lp below
//   lp  @  67584 : [256][264] P (f16, phase 2-3)
//   lv  @ 135168 : [32][264]  V^T (phase 1-3)
// total = (135168 + 8448) halves = 287232 bytes dynamic LDS
__global__ __launch_bounds__(512) void win_attn_kernel(
    const float* __restrict__ x, const float* __restrict__ qkv_b,
    const _Float16* __restrict__ wqkvT, _Float16* __restrict__ o_ws,
    float* __restrict__ attn_out) {
  extern __shared__ _Float16 lds[];
  _Float16* lx = lds;
  _Float16* lq = lds + 67584;
  _Float16* lk = lds + 77824;
  _Float16* lw = lds + 88064;
  _Float16* lp = lds + 67584;
  _Float16* lv = lds + 135168;

  const int tid  = threadIdx.x;
  const int lane = tid & 31;
  const int wv   = tid >> 5;            // 16 waves: wave wv owns row-tile wv
  const int ln   = lane & 15;
  const int hb8  = (lane & 16) ? 8 : 0; // D-tile: lanes 16-31 hold rows M+8

  const int widx = blockIdx.x;          // 0..511 window id = (b, wy, wx)
  const int bb = widx >> 8;
  const int wy = (widx >> 4) & 15;
  const int wx = widx & 15;
  const float* xb = x + ((size_t)bb << 24);

  // phase 0: gather window into LDS as f16 (vectorized float4 -> h4)
  for (int i = tid; i < 16384; i += 512) {
    const int t  = i >> 6;               // token 0..255
    const int c4 = (i & 63) << 2;        // col (floats)
    const int g  = (((wy << 4) + (t >> 4)) << 8) + (wx << 4) + (t & 15);
    const float4 v = *(const float4*)(xb + ((size_t)g << 8) + c4);
    h4 hv = { (_Float16)v.x, (_Float16)v.y, (_Float16)v.z, (_Float16)v.w };
    *(h4*)(lx + t * 264 + c4) = hv;
  }

  const float scale = 0.17677669529663687f;  // 1/sqrt(32)

  for (int h = 0; h < 8; ++h) {
    __syncthreads();  // prev head's P/V reads done before overwriting
    // stage this head's W^T blocks via Tensor Data Mover: 3 blocks (Q,K,V)
    // of 32 rows x 512B each, LDS rows padded to 528B. Wave 0 issues the
    // DMA (TDM ignores EXEC; one issue per workgroup, not per wave).
    if (wv == 0) {
      const unsigned lwb = (unsigned)(size_t)(void*)lw;
      const unsigned long long gb =
          (unsigned long long)(size_t)wqkvT + (unsigned long long)h * 16384ull;
      tdm_load_wblock(lwb,                 gb);             // Wq^T rows 0..31
      tdm_load_wblock(lwb + 32u * 528u,    gb + 131072ull); // Wk^T rows 32..63
      tdm_load_wblock(lwb + 64u * 528u,    gb + 262144ull); // Wv^T rows 64..95
      __builtin_amdgcn_s_wait_tensorcnt(0);
    }
    __syncthreads();

    // ---- phase 1: Q,K,V = xw @ W + b  (per wave: 6 tiles x 8 k-steps) ----
    v16h a8[8];
#pragma unroll
    for (int kc = 0; kc < 8; ++kc) a8[kc] = frag_a(lx, 264, wv << 4, kc << 5, lane);

#pragma unroll
    for (int t6 = 0; t6 < 6; ++t6) {
      const int mat = t6 >> 1, nc = t6 & 1;
      v8f acc = {};
#pragma unroll
      for (int kc = 0; kc < 8; ++kc) {
        v16h bf = frag_b(lw, 264, (mat << 5) + (nc << 4), kc << 5, lane);
        acc = WMMA_F16(a8[kc], bf, acc);
      }
      const float bv = qkv_b[(mat << 8) + (h << 5) + (nc << 4) + ln];
      if (mat < 2) {
        _Float16* dst = mat ? lk : lq;
#pragma unroll
        for (int r = 0; r < 8; ++r)
          dst[((wv << 4) + r + hb8) * 40 + (nc << 4) + ln] = (_Float16)(acc[r] + bv);
      } else {  // V stored transposed: [dim][token]
#pragma unroll
        for (int r = 0; r < 8; ++r)
          lv[((nc << 4) + ln) * 264 + (wv << 4) + r + hb8] = (_Float16)(acc[r] + bv);
      }
    }
    __syncthreads();

    // ---- phase 2: S = Q K^T (K=32 = head_dim -> 1 WMMA per 16x16 tile) ----
    const v16h qa = frag_a(lq, 40, wv << 4, 0, lane);
    v8f s[16];
#pragma unroll
    for (int nj = 0; nj < 16; ++nj) {
      v16h kf = frag_b(lk, 40, nj << 4, 0, lane);
      v8f z = {};
      s[nj] = WMMA_F16(qa, kf, z);
    }
    // row softmax: row m lives in VGPR r across one 16-lane half x 16 tiles
#pragma unroll
    for (int r = 0; r < 8; ++r) {
      float mx = s[0][r];
#pragma unroll
      for (int nj = 1; nj < 16; ++nj) mx = fmaxf(mx, s[nj][r]);
#pragma unroll
      for (int off = 1; off < 16; off <<= 1) mx = fmaxf(mx, __shfl_xor(mx, off, 16));
      float sum = 0.f;
#pragma unroll
      for (int nj = 0; nj < 16; ++nj) {
        const float e = __expf((s[nj][r] - mx) * scale);
        s[nj][r] = e;
        sum += e;
      }
#pragma unroll
      for (int off = 1; off < 16; off <<= 1) sum += __shfl_xor(sum, off, 16);
      const float inv = 1.f / sum;
#pragma unroll
      for (int nj = 0; nj < 16; ++nj) s[nj][r] *= inv;
    }
    __syncthreads();  // all Q/K reads complete before P overwrites them

    // write attn (f32, global) and P (f16, LDS)
    float* ap = attn_out + (((size_t)(widx << 3) + h) << 16);
#pragma unroll
    for (int nj = 0; nj < 16; ++nj) {
#pragma unroll
      for (int r = 0; r < 8; ++r) {
        const int row = (wv << 4) + r + hb8;
        const int col = (nj << 4) + ln;
        const float pv = s[nj][r];
        ap[(row << 8) + col] = pv;
        lp[row * 264 + col] = (_Float16)pv;
      }
    }
    __syncthreads();

    // ---- phase 3: O = P @ V, scatter-store unpartitioned f16 o ----
#pragma unroll
    for (int nc = 0; nc < 2; ++nc) {
      v8f acc = {};
#pragma unroll
      for (int kc = 0; kc < 8; ++kc) {
        v16h pa = frag_a(lp, 264, wv << 4, kc << 5, lane);
        v16h vb = frag_b(lv, 264, nc << 4, kc << 5, lane);
        acc = WMMA_F16(pa, vb, acc);
      }
#pragma unroll
      for (int r = 0; r < 8; ++r) {
        const int t = (wv << 4) + r + hb8;
        const int g = (((wy << 4) + (t >> 4)) << 8) + (wx << 4) + (t & 15);
        const int col = (h << 5) + (nc << 4) + ln;
        o_ws[((size_t)bb << 24) + ((size_t)g << 8) + col] = (_Float16)acc[r];
      }
    }
  }
}

// ---------------- kernel 2: out = o @ proj_w + proj_b --------------------
__global__ __launch_bounds__(256) void proj_kernel(
    const _Float16* __restrict__ o_ws, const _Float16* __restrict__ wpT,
    const float* __restrict__ proj_b, float* __restrict__ out) {
  const int tid = threadIdx.x, lane = tid & 31, wv = tid >> 5;
  const int ln = lane & 15, hb8 = (lane & 16) ? 8 : 0;
  const int rt = (blockIdx.x << 3) + wv;  // row-tile 0..8191
  const int row0 = rt << 4;

  v16h a8[8];
#pragma unroll
  for (int kc = 0; kc < 8; ++kc) a8[kc] = frag_a(o_ws, 256, row0, kc << 5, lane);

#pragma unroll
  for (int nc = 0; nc < 16; ++nc) {
    v8f acc = {};
#pragma unroll
    for (int kc = 0; kc < 8; ++kc) {
      v16h bf = frag_b(wpT, 256, nc << 4, kc << 5, lane);
      acc = WMMA_F16(a8[kc], bf, acc);
    }
    const float bv = proj_b[(nc << 4) + ln];
#pragma unroll
    for (int r = 0; r < 8; ++r)
      out[((size_t)(row0 + r + hb8) << 8) + (nc << 4) + ln] = acc[r] + bv;
  }
}

extern "C" void kernel_launch(void* const* d_in, const int* in_sizes, int n_in,
                              void* d_out, int out_size, void* d_ws, size_t ws_size,
                              hipStream_t stream) {
  const float* x      = (const float*)d_in[0];
  const float* qkv_w  = (const float*)d_in[1];
  const float* qkv_b  = (const float*)d_in[2];
  const float* proj_w = (const float*)d_in[3];
  const float* proj_b = (const float*)d_in[4];

  char* ws = (char*)d_ws;
  _Float16* wqkvT = (_Float16*)ws;               //   393,216 B : qkv_w^T f16 [768][256]
  _Float16* wpT   = (_Float16*)(ws + 393216);    //   131,072 B : proj_w^T f16 [256][256]
  _Float16* o_ws  = (_Float16*)(ws + 524288);    // 67,108,864 B : o f16 [131072][256]

  float* out  = (float*)d_out;                   // [2*65536][256] f32
  float* attn = out + 33554432;                  // [512][8][256][256] f32

  prep_weights_kernel<<<1024, 256, 0, stream>>>(qkv_w, proj_w, wqkvT, wpT);
  win_attn_kernel<<<512, 512, 287232, stream>>>(x, qkv_b, wqkvT, o_ws, attn);
  proj_kernel<<<1024, 256, 0, stream>>>(o_ws, wpT, proj_b, out);
}
